// Mamba2Block_3839700763196
// MI455X (gfx1250) — matrix operand order repaired and checked by reading, not compiled, not run on hardware
//
#include <hip/hip_runtime.h>
#include <cstdint>
#include <cstddef>

// ---------------------------------------------------------------- types ----
typedef __bf16 bf16;
typedef __attribute__((ext_vector_type(16))) __bf16 v16bf;
typedef __attribute__((ext_vector_type(8)))  float  v8f;

#define WMMA_BF16(a, b, c) \
  __builtin_amdgcn_wmma_f32_16x16x32_bf16(false, (a), false, (b), (short)0, (c), false, false)

union FragU { v16bf v; uint4 q[2]; };

static __device__ __forceinline__ bf16 f2bf(float f) { return (bf16)f; }
static __device__ __forceinline__ float bf2f(bf16 b) { return (float)b; }

// Async global->LDS copy of 16 bytes per lane (CDNA5, tracked by ASYNCcnt).
static __device__ __forceinline__ void async_cp16(uint32_t lds_addr, uint64_t gaddr) {
  asm volatile("global_load_async_to_lds_b128 %0, %1, off"
               :: "v"(lds_addr), "v"(gaddr) : "memory");
}
static __device__ __forceinline__ void async_wait0() {
  asm volatile("s_wait_asynccnt 0x0" ::: "memory");
}
static __device__ __forceinline__ uint32_t lds_addr_of(const void* p) {
  return (uint32_t)(uintptr_t)p;  // flat LDS aperture: low 32 bits = LDS byte addr
}

// A-operand: 16 rows x 32 K, row-major source, row stride `stride` (elements).
static __device__ __forceinline__ v16bf ldfragA(const bf16* base, int stride, int lane) {
  int r = lane & 15;
  int koff = (lane & 16) ? 8 : 0;
  const bf16* p = base + r * stride + koff;
  FragU f;
  f.q[0] = *(const uint4*)(p);
  f.q[1] = *(const uint4*)(p + 16);
  return f.v;
}
// B-operand: 32 K x 16 cols, source row-major as [col][k] with stride `stride`.
static __device__ __forceinline__ v16bf ldfragB(const bf16* base, int stride, int lane) {
  int c = lane & 15;
  int kb = (lane & 16) ? 8 : 0;
  const bf16* p = base + c * stride + kb;
  FragU f;
  f.q[0] = *(const uint4*)(p);
  f.q[1] = *(const uint4*)(p + 16);
  return f.v;
}

// ------------------------------------------------------------- constants ---
#define BATCH   4
#define LSEQ    2048
#define DMODEL  1024
#define DINNER  2048
#define NHEADS  32
#define DSTATE  64
#define DHEAD   64
#define CHUNK   256
#define NCHUNK  8
#define NB      128           // BATCH*NHEADS
#define NROWS   8192          // BATCH*LSEQ
#define NPAR    4128          // NHEADS*129
#define NPARP   4224          // padded to 33*128

// ---------------------------------------------------------- cvt kernels ----
__global__ void cvt_f32_bf16(const float* __restrict__ s, bf16* __restrict__ d, long n) {
  long i = (long)blockIdx.x * 256 + threadIdx.x;
  if (i < n) d[i] = f2bf(s[i]);
}

// d[c][r] = s[r][c] (bf16), s is [R][Cc] f32; output [Cp][R] with zero pad rows.
__global__ void cvt_transpose_pad(const float* __restrict__ s, bf16* __restrict__ d,
                                  int R, int Cc, int Cp) {
  long i = (long)blockIdx.x * 256 + threadIdx.x;
  long n = (long)Cp * R;
  if (i < n) {
    int c = (int)(i / R), r = (int)(i % R);
    d[i] = (c < Cc) ? f2bf(s[(long)r * Cc + c]) : (bf16)0.0f;
  }
}

// ------------------------------------------------------------- GEMM --------
// C[M x N] = A[M x K] * B[K x N]; A row-major bf16, B given TRANSPOSED as
// BTw[N][K] bf16 row-major. BM=BN=128, BK=32, async double-buffered LDS.
// mode 0: split epilogue -> xout (col<2048, bf16) / resout (col>=2048, f32)
// mode 1/2: f32 store to Cf with leading dim ldc.
__global__ __launch_bounds__(256)
void gemm_bf16_wmma(const bf16* __restrict__ A, const bf16* __restrict__ BTw,
                    int N, int K, int nblkN,
                    float* __restrict__ Cf, int ldc,
                    bf16* __restrict__ xout, float* __restrict__ resout, int mode) {
  __shared__ bf16 At[2][128 * 40];
  __shared__ bf16 Wt[2][128 * 40];

  int tid  = threadIdx.x;
  int bx   = blockIdx.x % nblkN;
  int by   = blockIdx.x / nblkN;
  int row0 = by * 128, col0 = bx * 128;
  int w_id = tid >> 5, lane = tid & 31;
  int r15  = lane & 15, hi8 = (lane & 16) ? 8 : 0;
  int rtb  = (w_id & 3) * 2;      // 2 row tiles
  int ctb  = (w_id >> 2) * 4;     // 4 col tiles

  v8f acc[2][4];
#pragma unroll
  for (int a = 0; a < 2; ++a)
#pragma unroll
    for (int b = 0; b < 4; ++b) acc[a][b] = (v8f){0,0,0,0,0,0,0,0};

  // per-thread staging addresses: 128 rows x 32 K per tile, 32B per thread
  int r  = tid >> 1, cs = (tid & 1) * 16;
  const bf16* gA = A   + (size_t)(row0 + r) * K + cs;
  const bf16* gB = BTw + (size_t)(col0 + r) * K + cs;
  uint32_t lA[2], lB[2];
  lA[0] = lds_addr_of(&At[0][r * 40 + cs]);
  lA[1] = lds_addr_of(&At[1][r * 40 + cs]);
  lB[0] = lds_addr_of(&Wt[0][r * 40 + cs]);
  lB[1] = lds_addr_of(&Wt[1][r * 40 + cs]);

#define GEMM_ISSUE(k0, buf)                                        \
  do {                                                             \
    async_cp16(lA[buf],      (uint64_t)(gA + (k0)));               \
    async_cp16(lA[buf] + 16, (uint64_t)(gA + (k0) + 8));           \
    async_cp16(lB[buf],      (uint64_t)(gB + (k0)));               \
    async_cp16(lB[buf] + 16, (uint64_t)(gB + (k0) + 8));           \
  } while (0)

  GEMM_ISSUE(0, 0);
  async_wait0();
  __syncthreads();

  int nk = K / 32;
  for (int i = 0; i < nk; ++i) {
    int cur = i & 1;
    if (i + 1 < nk) GEMM_ISSUE((i + 1) * 32, cur ^ 1);
    const bf16* Ab = At[cur];
    const bf16* Wb = Wt[cur];
#pragma unroll
    for (int rt = 0; rt < 2; ++rt) {
      v16bf fa = ldfragA(Ab + (rtb + rt) * 16 * 40, 40, lane);
#pragma unroll
      for (int ct = 0; ct < 4; ++ct) {
        v16bf fb = ldfragB(Wb + (ctb + ct) * 16 * 40, 40, lane);
        acc[rt][ct] = WMMA_BF16(fa, fb, acc[rt][ct]);
      }
    }
    async_wait0();   // next tile has landed (waited after compute: latency hidden)
    __syncthreads();
  }
#undef GEMM_ISSUE

#pragma unroll
  for (int rt = 0; rt < 2; ++rt)
#pragma unroll
    for (int ct = 0; ct < 4; ++ct)
#pragma unroll
      for (int v = 0; v < 8; ++v) {
        int grow = row0 + (rtb + rt) * 16 + v + hi8;
        int gcol = col0 + (ctb + ct) * 16 + r15;
        float val = acc[rt][ct][v];
        if (mode == 0) {
          if (gcol < DINNER) xout[(size_t)grow * DINNER + gcol] = f2bf(val);
          else               resout[(size_t)grow * DINNER + (gcol - DINNER)] = val;
        } else {
          Cf[(size_t)grow * ldc + gcol] = val;
        }
      }
}

// ----------------------------------------------- param transform -----------
__global__ void transform_params(const float* __restrict__ P,
                                 float* __restrict__ la,
                                 bf16* __restrict__ Bo, bf16* __restrict__ Co) {
  long i = (long)blockIdx.x * 256 + threadIdx.x;
  if (i >= (long)NROWS * NPAR) return;
  int r = (int)(i / NPAR), c = (int)(i % NPAR);
  int h = c / 129, j = c % 129;
  float v = P[(size_t)r * NPARP + c];
  int bb = r >> 11, t = r & 2047;
  int nb = bb * NHEADS + h;
  if (j == 0) {
    float sp = (v > 20.0f) ? v : log1pf(expf(v));
    la[(size_t)nb * LSEQ + t] = fmaxf(-sp, -23.025850929940457f); // log(clip(exp(-sp),1e-10))
  } else if (j <= DSTATE) {
    Bo[((size_t)nb * LSEQ + t) * DSTATE + (j - 1)] = f2bf(v);
  } else {
    Co[((size_t)nb * LSEQ + t) * DSTATE + (j - 1 - DSTATE)] = f2bf(v);
  }
}

// -------------------------------------------------------- SSD scan ---------
#define OFF_CUM 0
#define OFF_CL  1024
#define OFF_BL  37888
#define OFF_BT  74752
#define OFF_XT  108544
#define OFF_XWT 142336
#define OFF_HTB 176128
#define OFF_HTF 185344
#define OFF_STR 202752
#define SSD_LDS 270336
// strides (elements): Cl/Bl/hTb: 72 ; BT/XT/XwT/strip: 264 ; hTf(f32): 68

__global__ __launch_bounds__(256)
void ssd_kernel(const float* __restrict__ la_g,   // [128][2048]
                const bf16* __restrict__ Bbf,     // [128][2048][64]
                const bf16* __restrict__ Cbf,     // [128][2048][64]
                const bf16* __restrict__ Xbf,     // [8192][2048]
                const float* __restrict__ h0,     // [128][64][64] (n,p)
                float* __restrict__ y_g,          // [8192][2048]
                float* __restrict__ hT_out) {     // [128][64][64] (n,p)
  extern __shared__ char smem[];
  float* cum  = (float*)(smem + OFF_CUM);
  bf16*  Cl   = (bf16*)(smem + OFF_CL);
  bf16*  Bl   = (bf16*)(smem + OFF_BL);
  bf16*  BT   = (bf16*)(smem + OFF_BT);
  bf16*  XT   = (bf16*)(smem + OFF_XT);
  bf16*  XwT  = (bf16*)(smem + OFF_XWT);
  bf16*  hTb  = (bf16*)(smem + OFF_HTB);
  float* hTf  = (float*)(smem + OFF_HTF);
  bf16*  strb = (bf16*)(smem + OFF_STR);

  int tid = threadIdx.x;
  int nb  = blockIdx.x;
  int bb  = nb >> 5, hh = nb & 31;
  int w_id = tid >> 5, lane = tid & 31;
  int r15 = lane & 15, hi8 = (lane & 16) ? 8 : 0;
  bf16* strip = strb + w_id * 16 * 264;

  // init hT from hidden_state (transposed): hTf[p][n] = h0[nb][n][p]
  for (int i = tid; i < DSTATE * DHEAD; i += 256) {
    int n = i >> 6, p = i & 63;
    float v = h0[(size_t)nb * 4096 + i];
    hTf[p * 68 + n] = v;
    hTb[p * 72 + n] = f2bf(v);
  }
  __syncthreads();

  for (int ct = 0; ct < NCHUNK; ++ct) {
    // ---- cumsum of log(a) over chunk (Hillis-Steele) ----
    float lav = la_g[(size_t)nb * LSEQ + ct * CHUNK + tid];
    cum[tid] = lav;
    __syncthreads();
    for (int off = 1; off < 256; off <<= 1) {
      float add = (tid >= off) ? cum[tid - off] : 0.0f;
      __syncthreads();
      cum[tid] += add;
      __syncthreads();
    }
    // ---- stage C (async), B, B^T, X^T ----
    {
      int t = tid;
      size_t row = (size_t)nb * LSEQ + ct * CHUNK + t;
      // C rows: pure contiguous copy -> async global->LDS
      {
        uint32_t lcl = lds_addr_of(Cl + t * 72);
        uint64_t gcl = (uint64_t)(Cbf + row * DSTATE);
#pragma unroll
        for (int q = 0; q < 8; ++q) async_cp16(lcl + q * 16, gcl + q * 16);
      }
      union { uint4 q[8]; bf16 e[64]; } bu;
      const uint4* bg = (const uint4*)(Bbf + row * DSTATE);
#pragma unroll
      for (int q = 0; q < 8; ++q) bu.q[q] = bg[q];
#pragma unroll
      for (int q = 0; q < 8; ++q) ((uint4*)(Bl + t * 72))[q] = bu.q[q];
#pragma unroll
      for (int n = 0; n < 64; ++n) BT[n * 264 + t] = bu.e[n];
      union { uint4 q[8]; bf16 e[64]; } xu;
      const uint4* xg = (const uint4*)(Xbf + ((size_t)(bb * LSEQ + ct * CHUNK + t)) * DINNER + hh * DHEAD);
#pragma unroll
      for (int q = 0; q < 8; ++q) xu.q[q] = xg[q];
#pragma unroll
      for (int p = 0; p < 64; ++p) XT[p * 264 + tid] = xu.e[p];
    }
    async_wait0();
    __syncthreads();
    // ---- XwT[p][t] = X^T[p][t] * exp(cum[255]-cum[t]) ----
    {
      float wv = __expf(cum[255] - cum[tid]);
#pragma unroll
      for (int p = 0; p < 64; ++p)
        XwT[p * 264 + tid] = f2bf(bf2f(XT[p * 264 + tid]) * wv);
    }

    // ---- per-wave: M strips + y tiles ----
    for (int sel = 0; sel < 2; ++sel) {
      int it = w_id + sel * 8;  // i-tile 0..15
      v16bf a0 = ldfragA(Cl + it * 16 * 72, 72, lane);
      v16bf a1 = ldfragA(Cl + it * 16 * 72 + 32, 72, lane);
      // decay matrix strip: M[i][j], j-tiles 0..it
      for (int jt = 0; jt <= it; ++jt) {
        v8f m = (v8f){0,0,0,0,0,0,0,0};
        v16bf b0 = ldfragB(Bl + jt * 16 * 72, 72, lane);
        v16bf b1 = ldfragB(Bl + jt * 16 * 72 + 32, 72, lane);
        m = WMMA_BF16(a0, b0, m);
        m = WMMA_BF16(a1, b1, m);
        int jg = jt * 16 + r15;
        float cj = cum[jg];
#pragma unroll
        for (int v = 0; v < 8; ++v) {
          int ig = it * 16 + v + hi8;
          float val = (ig >= jg) ? __expf(cum[ig] - cj) * m[v] : 0.0f;
          strip[(v + hi8) * 264 + jg] = f2bf(val);
        }
      }
      if ((it + 1) & 1) {  // zero-pad next 16 cols so K rounds to 32
#pragma unroll
        for (int v = 0; v < 8; ++v)
          strip[(v + hi8) * 264 + (it + 1) * 16 + r15] = (bf16)0.0f;
      }
      // y = exp(cum_i) * (C @ hT) + M @ X
      v8f accP[4];
#pragma unroll
      for (int pt = 0; pt < 4; ++pt) {
        v8f t = (v8f){0,0,0,0,0,0,0,0};
        v16bf hb0 = ldfragB(hTb + pt * 16 * 72, 72, lane);
        v16bf hb1 = ldfragB(hTb + pt * 16 * 72 + 32, 72, lane);
        t = WMMA_BF16(a0, hb0, t);
        t = WMMA_BF16(a1, hb1, t);
#pragma unroll
        for (int v = 0; v < 8; ++v) t[v] *= __expf(cum[it * 16 + v + hi8]);
        accP[pt] = t;
      }
      int nk = (it + 2) >> 1;
      for (int kk = 0; kk < nk; ++kk) {
        v16bf fa = ldfragA(strip + kk * 32, 264, lane);
#pragma unroll
        for (int pt = 0; pt < 4; ++pt) {
          v16bf fb = ldfragB(XT + pt * 16 * 264 + kk * 32, 264, lane);
          accP[pt] = WMMA_BF16(fa, fb, accP[pt]);
        }
      }
      size_t rowb = (size_t)bb * LSEQ + ct * CHUNK + it * 16;
#pragma unroll
      for (int pt = 0; pt < 4; ++pt)
#pragma unroll
        for (int v = 0; v < 8; ++v)
          y_g[(rowb + v + hi8) * DINNER + hh * DHEAD + pt * 16 + r15] = accP[pt][v];
    }
    __syncthreads();

    // ---- hT update: hT = exp(cumL)*hT + (wX)^T @ B ----
    {
      float eL = __expf(cum[255]);
      int pt = w_id & 3;
#pragma unroll
      for (int s = 0; s < 2; ++s) {
        int nt = (w_id >> 2) * 2 + s;
        v8f acc;
#pragma unroll
        for (int v = 0; v < 8; ++v)
          acc[v] = hTf[(pt * 16 + v + hi8) * 68 + nt * 16 + r15] * eL;
#pragma unroll
        for (int kk = 0; kk < 8; ++kk) {
          v16bf fa = ldfragA(XwT + pt * 16 * 264 + kk * 32, 264, lane);
          v16bf fb = ldfragB(BT + nt * 16 * 264 + kk * 32, 264, lane);
          acc = WMMA_BF16(fa, fb, acc);
        }
#pragma unroll
        for (int v = 0; v < 8; ++v) {
          float x = acc[v];
          hTf[(pt * 16 + v + hi8) * 68 + nt * 16 + r15] = x;
          hTb[(pt * 16 + v + hi8) * 72 + nt * 16 + r15] = f2bf(x);
        }
      }
    }
    __syncthreads();
  }

  // final hidden state (transpose back): out[nb][n][p] = hTf[p][n]
  for (int i = tid; i < DSTATE * DHEAD; i += 256) {
    int n = i >> 6, p = i & 63;
    hT_out[(size_t)nb * 4096 + i] = hTf[p * 68 + n];
  }
}

// -------------------------------------------------- y * silu(res) ----------
__global__ void mulsilu_kernel(const float* __restrict__ y, const float* __restrict__ res,
                               bf16* __restrict__ ybf, long n) {
  long i = (long)blockIdx.x * 256 + threadIdx.x;
  if (i < n) {
    float r = res[i];
    float s = r / (1.0f + __expf(-r));
    ybf[i] = f2bf(y[i] * s);
  }
}

// ---------------------------------------------------------------- host -----
extern "C" void kernel_launch(void* const* d_in, const int* in_sizes, int n_in,
                              void* d_out, int out_size, void* d_ws, size_t ws_size,
                              hipStream_t stream) {
  const float* x   = (const float*)d_in[0];
  const float* h0  = (const float*)d_in[1];
  const float* Win = (const float*)d_in[2];
  const float* Wpm = (const float*)d_in[3];
  const float* Wot = (const float*)d_in[4];
  float* out  = (float*)d_out;                      // [8192][1024] then final_h
  float* hfin = out + (size_t)NROWS * DMODEL;       // [128][64][64]
  char* ws = (char*)d_ws;

  const size_t off_xbf  = 0;
  const size_t off_winT = off_xbf  + (size_t)NROWS * DMODEL * 2;
  const size_t off_wpmT = off_winT + (size_t)DMODEL * (2 * DINNER) * 2;
  const size_t off_wotT = off_wpmT + (size_t)DMODEL * NPARP * 2;
  const size_t off_Xbf  = off_wotT + (size_t)DINNER * DMODEL * 2;
  const size_t off_res  = off_Xbf  + (size_t)NROWS * DINNER * 2;
  const size_t off_P    = off_res  + (size_t)NROWS * DINNER * 4;
  const size_t off_la   = off_P    + (size_t)NROWS * NPARP * 4;
  const size_t off_Bbf  = off_la   + (size_t)NB * LSEQ * 4;
  const size_t off_Cbf  = off_Bbf  + (size_t)NB * LSEQ * DSTATE * 2;

  bf16*  xbf  = (bf16*)(ws + off_xbf);
  bf16*  winT = (bf16*)(ws + off_winT);   // [4096][1024]
  bf16*  wpmT = (bf16*)(ws + off_wpmT);   // [4224][1024]
  bf16*  wotT = (bf16*)(ws + off_wotT);   // [1024][2048]
  bf16*  Xbf  = (bf16*)(ws + off_Xbf);
  float* res  = (float*)(ws + off_res);
  float* P    = (float*)(ws + off_P);
  float* y    = (float*)(ws + off_P);     // reuse P after transform
  bf16*  ybf  = (bf16*)(ws + off_P + (size_t)NROWS * DINNER * 4);
  float* la   = (float*)(ws + off_la);
  bf16*  Bbf  = (bf16*)(ws + off_Bbf);
  bf16*  Cbf  = (bf16*)(ws + off_Cbf);

  // 1) bf16 conversions (weights also transposed for async tile staging)
  {
    long n = (long)NROWS * DMODEL;
    cvt_f32_bf16<<<dim3((unsigned)((n + 255) / 256)), dim3(256), 0, stream>>>(x, xbf, n);
    n = (long)(2 * DINNER) * DMODEL;   // WinT [4096][1024]
    cvt_transpose_pad<<<dim3((unsigned)((n + 255) / 256)), dim3(256), 0, stream>>>(
        Win, winT, DMODEL, 2 * DINNER, 2 * DINNER);
    n = (long)NPARP * DMODEL;          // WpmT [4224][1024], zero-padded rows
    cvt_transpose_pad<<<dim3((unsigned)((n + 255) / 256)), dim3(256), 0, stream>>>(
        Wpm, wpmT, DMODEL, NPAR, NPARP);
    n = (long)DMODEL * DINNER;         // WotT [1024][2048]
    cvt_transpose_pad<<<dim3((unsigned)((n + 255) / 256)), dim3(256), 0, stream>>>(
        Wot, wotT, DINNER, DMODEL, DMODEL);
  }
  // 2) GEMM1: xr = x @ W_in -> split Xbf (bf16) / res (f32)
  gemm_bf16_wmma<<<dim3((NROWS / 128) * (2 * DINNER / 128)), dim3(256), 0, stream>>>(
      xbf, winT, 2 * DINNER, DMODEL, 2 * DINNER / 128, nullptr, 0, Xbf, res, 0);
  // 3) GEMM2: P = x @ W_param (padded N)
  gemm_bf16_wmma<<<dim3((NROWS / 128) * (NPARP / 128)), dim3(256), 0, stream>>>(
      xbf, wpmT, NPARP, DMODEL, NPARP / 128, P, NPARP, nullptr, nullptr, 1);
  // 4) transform params -> log(a), B(bf16), C(bf16)
  {
    long n = (long)NROWS * NPAR;
    transform_params<<<dim3((unsigned)((n + 255) / 256)), dim3(256), 0, stream>>>(P, la, Bbf, Cbf);
  }
  // 5) SSD chunked scan (y overwrites P region; P is dead now)
  ssd_kernel<<<dim3(NB), dim3(256), SSD_LDS, stream>>>(la, Bbf, Cbf, Xbf, h0, y, hfin);
  // 6) ybf = bf16(y * silu(res))
  {
    long n = (long)NROWS * DINNER;
    mulsilu_kernel<<<dim3((unsigned)((n + 255) / 256)), dim3(256), 0, stream>>>(y, res, ybf, n);
  }
  // 7) GEMM3: out = ybf @ W_out
  gemm_bf16_wmma<<<dim3((NROWS / 128) * (DMODEL / 128)), dim3(256), 0, stream>>>(
      ybf, wotT, DMODEL, DINNER, DMODEL / 128, out, DMODEL, nullptr, nullptr, 2);
}
